// GraphAttentionLayer_10703058502073
// MI455X (gfx1250) — matrix operand order repaired and checked by reading, not compiled, not run on hardware
//
#include <hip/hip_runtime.h>
#include <hip/hip_bf16.h>
#include <math.h>
#include <stdint.h>

// Problem constants (from reference): B=8, N=1024, FIN=256, E=512, H=8, HD=64
#define NB   8
#define NN   1024
#define FIN  256
#define EE   512
#define NH   8
#define HD   64

typedef __attribute__((ext_vector_type(16))) __bf16 v16bf;
typedef __attribute__((ext_vector_type(8)))  float  v8f;

__device__ __forceinline__ __bf16 f2bf(float f) {
    union { float f; unsigned u; } v; v.f = f;
    unsigned r = v.u + 0x7FFFu + ((v.u >> 16) & 1u);   // round-to-nearest-even
    unsigned short h = (unsigned short)(r >> 16);
    __bf16 out;
    __builtin_memcpy(&out, &h, 2);
    return out;
}

__device__ __forceinline__ v8f wmma_bf16(v16bf a, v16bf b, v8f c) {
    // (neg_a, A, neg_b, B, c_mod, C, reuse_a, reuse_b)
    return __builtin_amdgcn_wmma_f32_16x16x32_bf16(false, a, false, b, (short)0, c, false, false);
}

// A-fragment (16x32 bf16, MxK): lane m = l&15; lanes 0-15 hold K {k..k+15},
// lanes 16-31 hold {k+16..k+31}  (contiguous order, matches B-fragments)
__device__ __forceinline__ v16bf load_a_frag(const __bf16* base, int ld, int row0, int k, int lane) {
    const int m  = lane & 15;
    const int hi = lane >> 4;
    const __bf16* p  = base + (size_t)(row0 + m) * ld + k;
    const __bf16* p0 = p + hi * 16;
    v16bf r;
#pragma unroll
    for (int i = 0; i < 16; ++i) r[i] = p0[i];
    return r;
}

// B-fragment (32x16 bf16, KxN) from a row-major (N x K) array (i.e. B^T storage):
// lane n = l&15 holds column n, 16 contiguous K values starting at k + (l>>4)*16
__device__ __forceinline__ v16bf load_b_frag(const __bf16* base, int ld, int col0, int k, int lane) {
    const int n  = lane & 15;
    const int hi = lane >> 4;
    const __bf16* p = base + (size_t)(col0 + n) * ld + k + hi * 16;
    v16bf r;
#pragma unroll
    for (int i = 0; i < 16; ++i) r[i] = p[i];
    return r;
}

// Same B-fragment layout but sourced from an LDS tile Bt[64][32]
__device__ __forceinline__ v16bf load_b_lds(const __bf16 (*Bt)[32], int col0, int lane) {
    const int n  = lane & 15;
    const int hi = lane >> 4;
    const __bf16* p = &Bt[col0 + n][hi * 16];
    v16bf r;
#pragma unroll
    for (int i = 0; i < 16; ++i) r[i] = p[i];
    return r;
}

// Async-copy one 64x32 bf16 B-tile (4 KB) from global to LDS.
// 128 threads x 2 chunks x 16 B. Tracked on ASYNCcnt (2 instructions per wave).
__device__ __forceinline__ void stage_b_async(const __bf16* __restrict__ Bw, int K, int col0, int k,
                                              __bf16 (*Bt)[32], int tid) {
#pragma unroll
    for (int i = 0; i < 2; ++i) {
        const int c     = tid + i * 128;       // chunk id 0..255
        const int row   = c >> 2;              // 0..63  (output column)
        const int off   = (c & 3) * 8;         // bf16 offset within the 32-wide K row
        const __bf16* g = Bw + (size_t)(col0 + row) * K + k + off;
        // Low 32 bits of a generic shared pointer are the wave-relative LDS byte address.
        const unsigned lds = (unsigned)(size_t)(const void*)&Bt[row][off];
        asm volatile("global_load_async_to_lds_b128 %0, %1, off"
                     :: "v"(lds), "v"(g)
                     : "memory");
    }
}

// ---------------------------------------------------------------------------
// f32 -> bf16 conversion
// ---------------------------------------------------------------------------
__global__ void cvt_f32_bf16(const float* __restrict__ in, __bf16* __restrict__ out, int n) {
    int i = blockIdx.x * blockDim.x + threadIdx.x;
    if (i < n) out[i] = f2bf(in[i]);
}

// ---------------------------------------------------------------------------
// Tiled WMMA GEMM: C(MxN) = A(MxK) @ Bw(NxK)^T + bias
// 128 threads (4 waves) per block -> 64x64 C tile; wave w owns rows w*16..w*16+15.
// B-tile (64x32) double-buffered in LDS via global_load_async_to_lds_b128;
// A-fragments register-prefetched one K-step ahead; B-frags batch-loaded from
// LDS so the 4 WMMAs issue back-to-back after a single DScnt wait.
// Output modes:
//   0: bf16, row-major (M x Nn)
//   1: bf16, head layout  idx = b*2^19 + h*2^16 + n*64 + d   (scramble AND (B,H,N,HD))
//   2: bf16, V-transposed idx = b*2^19 + h*2^16 + d*1024 + n (B,H,HD,N)
//   3: f32,  row-major (final output)
// ---------------------------------------------------------------------------
__global__ void __launch_bounds__(128)
gemm_wmma_bf16(const __bf16* __restrict__ A, const __bf16* __restrict__ Bw,
               const float* __restrict__ bias, void* __restrict__ Cout,
               int M, int Nn, int K, int mode) {
    __shared__ __bf16 Bt[2][64][32];

    const int tid  = threadIdx.x;
    const int lane = tid & 31;
    const int wave = tid >> 5;
    const int row0 = blockIdx.x * 64 + wave * 16;
    const int col0 = blockIdx.y * 64;

    v8f acc[4] = {v8f{}, v8f{}, v8f{}, v8f{}};

    // prologue: stage first B tile, load first A fragment
    stage_b_async(Bw, K, col0, 0, Bt[0], tid);
    v16bf aA = load_a_frag(A, K, row0, 0, lane);

    const int KT = K >> 5;
    for (int kt = 0; kt < KT; ++kt) {
        const int cur = kt & 1;
        const bool more = (kt + 1) < KT;
        if (more) stage_b_async(Bw, K, col0, (kt + 1) * 32, Bt[cur ^ 1], tid);  // overwrite guarded by prev barrier
        v16bf aN = aA;
        if (more) aN = load_a_frag(A, K, row0, (kt + 1) * 32, lane);            // in flight during WMMAs
        if (more) asm volatile("s_wait_asynccnt 0x2" ::: "memory");             // stage kt retired
        else      asm volatile("s_wait_asynccnt 0x0" ::: "memory");
        __syncthreads();                                                        // all waves' stages visible
        v16bf bb[4];
#pragma unroll
        for (int c = 0; c < 4; ++c) bb[c] = load_b_lds(Bt[cur], c * 16, lane);  // batch DS loads
#pragma unroll
        for (int c = 0; c < 4; ++c) acc[c] = wmma_bf16(aA, bb[c], acc[c]);      // back-to-back XDL
        __syncthreads();                                                        // done reading before overwrite
        aA = aN;
    }

    const int hi = lane >> 4, lo = lane & 15;
#pragma unroll
    for (int c = 0; c < 4; ++c) {
        const int col = col0 + c * 16 + lo;
        const float bv = bias ? bias[col] : 0.0f;
#pragma unroll
        for (int r = 0; r < 8; ++r) {
            const int row = row0 + hi * 8 + r;
            const float val = acc[c][r] + bv;
            if (mode == 3) {
                ((float*)Cout)[(size_t)row * Nn + col] = val;
            } else {
                size_t idx;
                if (mode == 0) {
                    idx = (size_t)row * Nn + col;
                } else {
                    const int b = row >> 10, n = row & 1023;
                    const int h = col >> 6,  d = col & 63;
                    if (mode == 1) idx = ((size_t)b << 19) + ((size_t)h << 16) + (size_t)n * 64 + d;
                    else           idx = ((size_t)b << 19) + ((size_t)h << 16) + (size_t)d * 1024 + n;
                }
                ((__bf16*)Cout)[idx] = f2bf(val);
            }
        }
    }
}

// ---------------------------------------------------------------------------
// Masked multi-head attention, flash-style online softmax.
// One wave per (b,h, 16-query tile). qp/kp: (B,H,N,HD) bf16; vpt: (B,H,HD,N) bf16.
// ctx written as (B,N,E) bf16.
// ---------------------------------------------------------------------------
__global__ void attn_flash_wmma(const __bf16* __restrict__ qp, const __bf16* __restrict__ kp,
                                const __bf16* __restrict__ vpt, const float* __restrict__ adj,
                                __bf16* __restrict__ ctx) {
    __shared__ __bf16 pT[16][32];

    const int lane = threadIdx.x;
    const int lo = lane & 15, hi = lane >> 4;
    const int bh = blockIdx.y;            // 0..63
    const int b = bh >> 3, h = bh & 7;
    const int q0 = blockIdx.x * 16;

    const __bf16* qbase = qp  + (size_t)bh * NN * HD;
    const __bf16* kbase = kp  + (size_t)bh * NN * HD;
    const __bf16* vbase = vpt + (size_t)bh * HD * NN;   // (HD, N)
    const float*  adjb  = adj + (size_t)b * NN * NN;

    // Q fragments, loaded once (HD = 64 -> two K=32 A-frags)
    const v16bf qa0 = load_a_frag(qbase, HD, q0, 0,  lane);
    const v16bf qa1 = load_a_frag(qbase, HD, q0, 32, lane);

    v8f O[4] = {v8f{}, v8f{}, v8f{}, v8f{}};
    float mrow[8], lrow[8];
#pragma unroll
    for (int r = 0; r < 8; ++r) { mrow[r] = -1e30f; lrow[r] = 0.0f; }

    for (int jj = 0; jj < NN; jj += 32) {
        // --- adjacency mask values: issue loads BEFORE score WMMAs (independent) ---
        const int k0 = jj + lo, k1 = jj + 16 + lo;
        float e0[8], e1[8];
#pragma unroll
        for (int r = 0; r < 8; ++r) {
            const int qg = q0 + hi * 8 + r;
            e0[r] = adjb[(size_t)qg * NN + k0] + ((qg == k0) ? 1.0f : 0.0f);
            e1[r] = adjb[(size_t)qg * NN + k1] + ((qg == k1) ? 1.0f : 0.0f);
        }

        // --- scores: batch-load 4 K-fragments, then 4 back-to-back WMMAs ---
        v16bf kb00 = load_b_frag(kbase, HD, jj,      0,  lane);
        v16bf kb01 = load_b_frag(kbase, HD, jj,      32, lane);
        v16bf kb10 = load_b_frag(kbase, HD, jj + 16, 0,  lane);
        v16bf kb11 = load_b_frag(kbase, HD, jj + 16, 32, lane);
        v8f s0 = v8f{}, s1 = v8f{};
        s0 = wmma_bf16(qa0, kb00, s0);
        s0 = wmma_bf16(qa1, kb01, s0);
        s1 = wmma_bf16(qa0, kb10, s1);
        s1 = wmma_bf16(qa1, kb11, s1);

        // --- scale + mask (mask = (adj + I) == 0 -> -inf) ---
#pragma unroll
        for (int r = 0; r < 8; ++r) {
            s0[r] = (e0[r] == 0.0f) ? -1e30f : s0[r] * 0.125f;   // 1/sqrt(64)
            s1[r] = (e1[r] == 0.0f) ? -1e30f : s1[r] * 0.125f;
        }

        // --- prefetch V fragments for this key block (independent of softmax) ---
        v16bf vb[4];
#pragma unroll
        for (int c = 0; c < 4; ++c)
            vb[c] = load_b_frag(vbase, NN, c * 16, jj, lane);    // col = hd, K = keys

        // --- online softmax (row lives across a 16-lane half: butterfly reduce) ---
#pragma unroll
        for (int r = 0; r < 8; ++r) {
            float v = fmaxf(s0[r], s1[r]);
#pragma unroll
            for (int off = 1; off < 16; off <<= 1)
                v = fmaxf(v, __shfl_xor(v, off, 32));
            const float mnew  = fmaxf(mrow[r], v);
            const float alpha = __expf(mrow[r] - mnew);
            const float p0 = __expf(s0[r] - mnew);
            const float p1 = __expf(s1[r] - mnew);
            float rs = p0 + p1;
#pragma unroll
            for (int off = 1; off < 16; off <<= 1)
                rs += __shfl_xor(rs, off, 32);
            lrow[r] = lrow[r] * alpha + rs;
            mrow[r] = mnew;
#pragma unroll
            for (int c = 0; c < 4; ++c) O[c][r] *= alpha;
            pT[hi * 8 + r][lo]      = f2bf(p0);
            pT[hi * 8 + r][16 + lo] = f2bf(p1);
        }
        __syncthreads();

        // --- P(16x32) @ V(32x64): restage P via LDS into A-frag layout ---
        const v16bf pa = load_a_frag(&pT[0][0], 32, 0, 0, lane);
        __syncthreads();
#pragma unroll
        for (int c = 0; c < 4; ++c)
            O[c] = wmma_bf16(pa, vb[c], O[c]);
    }

    // --- epilogue: normalize, store ctx as (B,N,E) bf16 ---
#pragma unroll
    for (int r = 0; r < 8; ++r) {
        const float inv = 1.0f / lrow[r];
        const int qg = q0 + hi * 8 + r;
#pragma unroll
        for (int c = 0; c < 4; ++c) {
            const size_t idx = ((size_t)(b * NN + qg)) * EE + h * HD + c * 16 + lo;
            ctx[idx] = f2bf(O[c][r] * inv);
        }
    }
}

// ---------------------------------------------------------------------------
// Host-side orchestration
// ---------------------------------------------------------------------------
extern "C" void kernel_launch(void* const* d_in, const int* in_sizes, int n_in,
                              void* d_out, int out_size, void* d_ws, size_t ws_size,
                              hipStream_t stream) {
    const float* x        = (const float*)d_in[0];   // (B,N,FIN)
    const float* adj      = (const float*)d_in[1];   // (B,N,N)
    const float* Wq       = (const float*)d_in[2];   // (E,FIN)
    const float* Wk       = (const float*)d_in[3];
    const float* Wv       = (const float*)d_in[4];
    const float* in_proj  = (const float*)d_in[5];   // (3E,E)
    const float* in_b     = (const float*)d_in[6];   // (3E)
    const float* mha_w    = (const float*)d_in[7];   // (E,E)
    const float* mha_b    = (const float*)d_in[8];
    const float* out_w    = (const float*)d_in[9];
    const float* out_b    = (const float*)d_in[10];
    float* out = (float*)d_out;                      // (B,N,E) f32

    char* ws = (char*)d_ws;
    const size_t M  = (size_t)NB * NN;               // 8192 rows
    size_t off = 0;
    __bf16* Xbf   = (__bf16*)(ws + off); off += (size_t)NB * NN * FIN * 2;   // 4 MB
    __bf16* Wqbf  = (__bf16*)(ws + off); off += (size_t)EE * FIN * 2;
    __bf16* Wkbf  = (__bf16*)(ws + off); off += (size_t)EE * FIN * 2;
    __bf16* Wvbf  = (__bf16*)(ws + off); off += (size_t)EE * FIN * 2;
    __bf16* IPbf  = (__bf16*)(ws + off); off += (size_t)3 * EE * EE * 2;
    __bf16* MHbf  = (__bf16*)(ws + off); off += (size_t)EE * EE * 2;
    __bf16* OWbf  = (__bf16*)(ws + off); off += (size_t)EE * EE * 2;
    __bf16* q_s   = (__bf16*)(ws + off); size_t off_q = off; off += M * EE * 2;  // 8 MB each
    __bf16* k_s   = (__bf16*)(ws + off); size_t off_k = off; off += M * EE * 2;
    __bf16* v_s   = (__bf16*)(ws + off); off += M * EE * 2;
    __bf16* qpB   = (__bf16*)(ws + off); off += M * EE * 2;
    __bf16* kpB   = (__bf16*)(ws + off); off += M * EE * 2;
    __bf16* vptB  = (__bf16*)(ws + off); off += M * EE * 2;
    __bf16* ctxB  = (__bf16*)(ws + off_q);   // reuse q_s (dead after proj2)
    __bf16* attB  = (__bf16*)(ws + off_k);   // reuse k_s
    (void)ws_size; (void)n_in; (void)in_sizes; (void)out_size;

    // 1) convert inputs/weights to bf16
    auto cvt = [&](const float* src, __bf16* dst, int n) {
        cvt_f32_bf16<<<(n + 255) / 256, 256, 0, stream>>>(src, dst, n);
    };
    cvt(x,       Xbf,  NB * NN * FIN);
    cvt(Wq,      Wqbf, EE * FIN);
    cvt(Wk,      Wkbf, EE * FIN);
    cvt(Wv,      Wvbf, EE * FIN);
    cvt(in_proj, IPbf, 3 * EE * EE);
    cvt(mha_w,   MHbf, EE * EE);
    cvt(out_w,   OWbf, EE * EE);

    const dim3 gblk(128);                       // 4 waves per block, 64x64 tile
    const dim3 ggrid(M / 64, EE / 64);
    // 2) first projections + scramble (mode 1): q/k/v = X @ W^T
    gemm_wmma_bf16<<<ggrid, gblk, 0, stream>>>(Xbf, Wqbf, nullptr, q_s, (int)M, EE, FIN, 1);
    gemm_wmma_bf16<<<ggrid, gblk, 0, stream>>>(Xbf, Wkbf, nullptr, k_s, (int)M, EE, FIN, 1);
    gemm_wmma_bf16<<<ggrid, gblk, 0, stream>>>(Xbf, Wvbf, nullptr, v_s, (int)M, EE, FIN, 1);
    // 3) second projections: qp/kp -> head layout (mode 1), vp -> transposed (mode 2)
    gemm_wmma_bf16<<<ggrid, gblk, 0, stream>>>(q_s, IPbf,                   in_b,          qpB,  (int)M, EE, EE, 1);
    gemm_wmma_bf16<<<ggrid, gblk, 0, stream>>>(k_s, IPbf + (size_t)EE*EE,   in_b + EE,     kpB,  (int)M, EE, EE, 1);
    gemm_wmma_bf16<<<ggrid, gblk, 0, stream>>>(v_s, IPbf + (size_t)2*EE*EE, in_b + 2*EE,   vptB, (int)M, EE, EE, 2);
    // 4) masked flash attention -> ctx (B,N,E) bf16
    {
        dim3 g(NN / 16, NB * NH);
        attn_flash_wmma<<<g, dim3(32), 0, stream>>>(qpB, kpB, vptB, adj, ctxB);
    }
    // 5) attended = ctx @ mha_out_w^T + b (bf16, mode 0)
    gemm_wmma_bf16<<<ggrid, gblk, 0, stream>>>(ctxB, MHbf, mha_b, attB, (int)M, EE, EE, 0);
    // 6) out = attended @ out_w^T + b (f32, mode 3)
    gemm_wmma_bf16<<<ggrid, gblk, 0, stream>>>(attB, OWbf, out_b, out, (int)M, EE, EE, 3);
}
